// GSAGELayer_2551210573908
// MI455X (gfx1250) — matrix operand order repaired and checked by reading, not compiled, not run on hardware
//
#include <hip/hip_runtime.h>

#define Bn 4
#define Nn 256
#define Ln 4
#define Hn 256
#define Dn 128
#define DEGn 16
#define Tn 64
#define ROWS 1024   // B*N
#define SEQn 4096   // B*N*L
#define G4H 1024    // 4*H
#define MT 32       // sequences per WG in the persistent LSTM

typedef __attribute__((ext_vector_type(16))) __bf16 v16bf;
typedef __attribute__((ext_vector_type(8)))  float  v8f;

__device__ __forceinline__ unsigned short f2bf(float f) {
    unsigned int u = __float_as_uint(f);
    unsigned int r = u + 0x7FFFu + ((u >> 16) & 1u);   // round-to-nearest-even
    return (unsigned short)(r >> 16);
}
__device__ __forceinline__ float sigm(float x) { return 1.0f / (1.0f + expf(-x)); }

// ---------------------------------------------------------------------------
// h = tanh(LayerNorm(concat(x, feature) @ fc_w^T + fc_b))
// ---------------------------------------------------------------------------
__global__ void __launch_bounds__(256)
fc_ln_tanh_kernel(const float* __restrict__ x, const float* __restrict__ feat,
                  const float* __restrict__ fc_w, const float* __restrict__ fc_b,
                  const float* __restrict__ ln_g, const float* __restrict__ ln_b,
                  float* __restrict__ hfc, unsigned short* __restrict__ hfcpack) {
    __shared__ float srow[272];
    __shared__ float hrow[256];
    __shared__ float psum[32], psq[32];
    int row = blockIdx.x;      // b*N + n
    int tid = threadIdx.x;     // 0..255
    srow[tid] = x[(size_t)row * 256 + tid];
    if (tid < 2) srow[256 + tid] = feat[(size_t)row * 2 + tid];
    __syncthreads();

    const float* wr = fc_w + (size_t)tid * 258;
    float acc = fc_b[tid];
    for (int k = 0; k < 258; k++) acc += srow[k] * wr[k];
    hrow[tid] = acc;
    __syncthreads();

    if (tid < 32) {
        float s = 0.f, q = 0.f;
        for (int k = 0; k < 8; k++) { float v = hrow[tid + 32 * k]; s += v; q += v * v; }
        psum[tid] = s; psq[tid] = q;
    }
    __syncthreads();
    float s = 0.f, q = 0.f;
    for (int k = 0; k < 32; k++) { s += psum[k]; q += psq[k]; }
    float mu  = s * (1.0f / 256.0f);
    float var = q * (1.0f / 256.0f) - mu * mu;
    float v = (acc - mu) * rsqrtf(var + 1e-5f) * ln_g[tid] + ln_b[tid];
    v = tanhf(v);
    hfc[(size_t)row * 256 + tid] = v;

    int rt = row >> 4, M = row & 15;
    int kc = tid >> 5, r = tid & 31;
    int hi = (r >> 3) & 1;
    int e  = (r & 7) + ((r & 16) ? 8 : 0);
    int lane = hi * 16 + M;
    hfcpack[((((size_t)rt * 8 + kc) * 32 + lane) << 4) + e] = f2bf(v);
}

// ---------------------------------------------------------------------------
// Pack w_hh^T into B-matrix fragment layout (bf16) + fold bias = b_ih + b_hh.
// ---------------------------------------------------------------------------
__global__ void pack_whh_kernel(const float* __restrict__ w_hh,
                                const float* __restrict__ b_ih,
                                const float* __restrict__ b_hh,
                                unsigned short* __restrict__ whhpack,
                                float* __restrict__ bias) {
    int gid = blockIdx.x * blockDim.x + threadIdx.x;
    if (gid < G4H) bias[gid] = b_ih[gid] + b_hh[gid];
    if (gid >= 64 * 8 * 32 * 16) return;
    int e = gid & 15, lane = (gid >> 4) & 31, kc = (gid >> 9) & 7, nt = gid >> 12;
    int n = nt * 16 + (lane & 15);
    int k = kc * 32 + ((e & 8) << 1) + ((lane >> 4) << 3) + (e & 7);
    whhpack[gid] = f2bf(w_hh[(size_t)n * 256 + k]);
}

// ---------------------------------------------------------------------------
// WIH[l] = W[l] @ w_ih^T folded GEMM, written directly in B-fragment layout.
// ---------------------------------------------------------------------------
__global__ void pack_wih_kernel(const float* __restrict__ W,
                                const float* __restrict__ w_ih,
                                unsigned short* __restrict__ wihpack) {
    int gid = blockIdx.x * blockDim.x + threadIdx.x;
    if (gid >= Ln * 64 * 8 * 32 * 16) return;
    int e = gid & 15, lane = (gid >> 4) & 31, kc = (gid >> 9) & 7;
    int nt = (gid >> 12) & 63, l = gid >> 18;
    int n = nt * 16 + (lane & 15);
    int k = kc * 32 + ((e & 8) << 1) + ((lane >> 4) << 3) + (e & 7);
    const float* wl = W + ((size_t)l * 256 + k) * 128;
    const float* wi = w_ih + (size_t)n * 128;
    float acc = 0.f;
    for (int d = 0; d < 128; d++) acc += wl[d] * wi[d];
    wihpack[gid] = f2bf(acc);
}

// ---------------------------------------------------------------------------
// Neighbor indices: adj[b,l,n,:] has exactly DEG ones; collect ascending.
// ---------------------------------------------------------------------------
__global__ void build_idx_kernel(const float* __restrict__ adj,
                                 int* __restrict__ idxout) {
    int gid = blockIdx.x * blockDim.x + threadIdx.x;    // (b*L + l)*N + n
    if (gid >= Bn * Ln * Nn) return;
    const float* rowp = adj + (size_t)gid * Nn;
    int cnt = 0;
    for (int j = 0; j < Nn && cnt < DEGn; j++)
        if (rowp[j] > 0.5f) idxout[gid * DEGn + cnt++] = j;
    for (; cnt < DEGn; cnt++) idxout[gid * DEGn + cnt] = 0;
}

// ---------------------------------------------------------------------------
// P[b,l,n,:] = h[b,n,:] @ WIH[l]   (WMMA bf16, fp32 accumulate)
// ---------------------------------------------------------------------------
__global__ void __launch_bounds__(256, 1)
pmat_kernel(const unsigned short* __restrict__ hfcpack,
            const unsigned short* __restrict__ wihpack,
            float* __restrict__ P) {
    int rt = blockIdx.x;
    int l  = blockIdx.y;
    int w    = threadIdx.x >> 5;
    int lane = threadIdx.x & 31;
    const v16bf* Ap = (const v16bf*)hfcpack;
    const v16bf* Bp = (const v16bf*)wihpack;
    v16bf afr[8];
#pragma unroll
    for (int kc = 0; kc < 8; kc++) afr[kc] = Ap[((size_t)rt * 8 + kc) * 32 + lane];
#pragma unroll
    for (int nl = 0; nl < 8; nl++) {
        int nt = w * 8 + nl;
        size_t bbase = (((size_t)l * 64 + nt) * 8) * 32 + lane;
        v16bf bfr[8];
#pragma unroll
        for (int kc = 0; kc < 8; kc++) bfr[kc] = Bp[bbase + (size_t)kc * 32];
        v8f acc0 = {0.f, 0.f, 0.f, 0.f, 0.f, 0.f, 0.f, 0.f};
        v8f acc1 = {0.f, 0.f, 0.f, 0.f, 0.f, 0.f, 0.f, 0.f};
#pragma unroll
        for (int kc = 0; kc < 8; kc += 2) {
            acc0 = __builtin_amdgcn_wmma_f32_16x16x32_bf16(
                false, afr[kc],     false, bfr[kc],     (short)0, acc0, false, false);
            acc1 = __builtin_amdgcn_wmma_f32_16x16x32_bf16(
                false, afr[kc + 1], false, bfr[kc + 1], (short)0, acc1, false, false);
        }
        int col   = nt * 16 + (lane & 15);
        int rbase = rt * 16 + ((lane >> 4) << 3);
#pragma unroll
        for (int r = 0; r < 8; r++) {
            int gr = rbase + r;
            int b = gr >> 8, n = gr & 255;
            P[(size_t)((b * 4 + l) * 256 + n) * G4H + col] = acc0[r] + acc1[r];
        }
    }
}

// ---------------------------------------------------------------------------
// Persistent full-sequence LSTM: each WG owns 32 sequences for all 64 steps.
//   - c state: 32 registers/thread; h tile: 16 KB LDS (bf16 A-fragment layout)
//   - gate tile: 128 KB LDS; total 144 KB/WG (CDNA5 WGP budget)
//   - per step: [32x1024] GEMM = 1024 wmma/WG; B (packed w_hh, L2-resident)
//     streamed with a 2-deep software pipeline; two independent accumulator
//     chains (one per m-tile) share each B fragment.
// ---------------------------------------------------------------------------
__global__ void __launch_bounds__(256, 1)
lstm_seq_kernel(const unsigned short* __restrict__ whhpack,
                const float* __restrict__ bias,
                const float* __restrict__ P,
                const int* __restrict__ idxb,
                float* __restrict__ hstate) {
    __shared__ float gsm[MT * G4H];            // 128 KB gate tile
    __shared__ unsigned short hp[MT * Hn];     // 16 KB packed h tile
    int st   = blockIdx.x;                     // sequence tile (32 seqs)
    int tid  = threadIdx.x;
    int w    = tid >> 5;
    int lane = tid & 31;
    int u    = tid;                            // hidden unit for update phase

    for (int i = tid; i < MT * Hn; i += 256) hp[i] = 0;
    float creg[MT];
#pragma unroll
    for (int r = 0; r < MT; r++) creg[r] = 0.f;
    float bi = bias[u], bf = bias[256 + u], bg = bias[512 + u], bo = bias[768 + u];

    const v16bf* Bp = (const v16bf*)whhpack;
    const v16bf* Al = (const v16bf*)hp;
    __syncthreads();

#pragma unroll 1
    for (int t = 0; t < Tn; t++) {
        // prefetch gathered P rows (4 KB each) behind the GEMM
        if (t < DEGn) {
            int row  = tid >> 3;               // 32 rows, 8 threads each
            int part = tid & 7;
            int s = st * MT + row;
            int b = s >> 10, n = (s >> 2) & 255, l = s & 3;
            int node = idxb[((b * 4 + l) * 256 + n) * DEGn + t];
            const char* pp = (const char*)(P + (size_t)((b * 4 + l) * 256 + node) * G4H);
            __builtin_prefetch(pp + part * 512, 0, 3);
            __builtin_prefetch(pp + part * 512 + 128, 0, 3);
            __builtin_prefetch(pp + part * 512 + 256, 0, 3);
            __builtin_prefetch(pp + part * 512 + 384, 0, 3);
        }

        // A fragments for both m-tiles from LDS
        v16bf afr0[8], afr1[8];
#pragma unroll
        for (int kc = 0; kc < 8; kc++) {
            afr0[kc] = Al[kc * 32 + lane];
            afr1[kc] = Al[(8 + kc) * 32 + lane];
        }

#pragma unroll
        for (int nl = 0; nl < 8; nl++) {
            int nt = w * 8 + nl;
            size_t bbase = ((size_t)nt * 8) * 32 + lane;
            v16bf bcur = Bp[bbase];
            v8f accA = {0.f, 0.f, 0.f, 0.f, 0.f, 0.f, 0.f, 0.f};
            v8f accB = {0.f, 0.f, 0.f, 0.f, 0.f, 0.f, 0.f, 0.f};
#pragma unroll
            for (int kc = 0; kc < 8; kc++) {
                v16bf bnxt = bcur;
                if (kc < 7) bnxt = Bp[bbase + (size_t)(kc + 1) * 32];
                accA = __builtin_amdgcn_wmma_f32_16x16x32_bf16(
                    false, afr0[kc], false, bcur, (short)0, accA, false, false);
                accB = __builtin_amdgcn_wmma_f32_16x16x32_bf16(
                    false, afr1[kc], false, bcur, (short)0, accB, false, false);
                bcur = bnxt;
            }
            int col = nt * 16 + (lane & 15);
            int rb  = (lane >> 4) << 3;
#pragma unroll
            for (int r = 0; r < 8; r++) {
                gsm[(rb + r) * G4H + col]        = accA[r];
                gsm[(16 + rb + r) * G4H + col]   = accB[r];
            }
        }
        __syncthreads();   // gates visible; all waves done reading hp

        // fused gate nonlinearity + state update; repack h into LDS
#pragma unroll 1
        for (int row = 0; row < MT; row++) {
            int s = st * MT + row;             // seq = (b*N + n)*L + l
            float gi = gsm[row * G4H + u]       + bi;
            float gf = gsm[row * G4H + 256 + u] + bf;
            float gg = gsm[row * G4H + 512 + u] + bg;
            float go = gsm[row * G4H + 768 + u] + bo;
            if (t < DEGn) {
                int b = s >> 10, n = (s >> 2) & 255, l = s & 3;
                int node = idxb[((b * 4 + l) * 256 + n) * DEGn + t];
                const float* pp = P + (size_t)((b * 4 + l) * 256 + node) * G4H;
                gi += pp[u]; gf += pp[256 + u]; gg += pp[512 + u]; go += pp[768 + u];
            }
            float cn = sigm(gf) * creg[row] + sigm(gi) * tanhf(gg);
            float hn = sigm(go) * tanhf(cn);
            creg[row] = cn;
            // repack into A-fragment layout for next step
            int m = row >> 4, row16 = row & 15;
            int kc = u >> 5, r = u & 31;
            int hi = (r >> 3) & 1;
            int e  = (r & 7) + ((r & 16) ? 8 : 0);
            int lane2 = hi * 16 + row16;
            hp[(((m * 8 + kc) * 32 + lane2) << 4) + e] = f2bf(hn);
            if (t == Tn - 1) hstate[(size_t)s * 256 + u] = hn;
        }
        __syncthreads();   // hp updates visible before next step's A loads
    }
}

// ---------------------------------------------------------------------------
// out[b,n,u] = 2*h[b,n,u] + mean_l last_hidden[(b,n,l)][u]
// ---------------------------------------------------------------------------
__global__ void final_kernel(const float* __restrict__ hfc,
                             const float* __restrict__ hstate,
                             float* __restrict__ out) {
    int gid = blockIdx.x * blockDim.x + threadIdx.x;
    if (gid >= ROWS * Hn) return;
    int u = gid & 255, rn = gid >> 8;
    size_t base = (size_t)rn * 4 * 256 + u;
    float sum = hstate[base] + hstate[base + 256] + hstate[base + 512] + hstate[base + 768];
    out[gid] = 2.0f * hfc[gid] + 0.25f * sum;
}

// ---------------------------------------------------------------------------
extern "C" void kernel_launch(void* const* d_in, const int* in_sizes, int n_in,
                              void* d_out, int out_size, void* d_ws, size_t ws_size,
                              hipStream_t stream) {
    (void)in_sizes; (void)n_in; (void)out_size; (void)ws_size;
    const float* x    = (const float*)d_in[0];
    const float* feat = (const float*)d_in[1];
    const float* adj  = (const float*)d_in[2];
    const float* fc_w = (const float*)d_in[3];
    const float* fc_b = (const float*)d_in[4];
    const float* ln_g = (const float*)d_in[5];
    const float* ln_b = (const float*)d_in[6];
    const float* W    = (const float*)d_in[7];
    const float* w_ih = (const float*)d_in[8];
    const float* w_hh = (const float*)d_in[9];
    const float* b_ih = (const float*)d_in[10];
    const float* b_hh = (const float*)d_in[11];

    char* base = (char*)d_ws;
    size_t off = 0;
    auto carve = [&](size_t bytes) -> char* {
        char* p = base + off;
        off += (bytes + 255) & ~(size_t)255;
        return p;
    };
    float*          hfc     = (float*)         carve((size_t)ROWS * Hn * 4);
    unsigned short* hfcpack = (unsigned short*)carve((size_t)ROWS * Hn * 2);
    unsigned short* whhpack = (unsigned short*)carve((size_t)G4H * Hn * 2);
    float*          bias    = (float*)         carve((size_t)G4H * 4);
    unsigned short* wihpack = (unsigned short*)carve((size_t)Ln * Hn * G4H * 2);
    float*          P       = (float*)         carve((size_t)Bn * Ln * Nn * G4H * 4);
    int*            idxb    = (int*)           carve((size_t)Bn * Ln * Nn * DEGn * 4);
    float*          hstate  = (float*)         carve((size_t)SEQn * Hn * 4);

    fc_ln_tanh_kernel<<<ROWS, 256, 0, stream>>>(x, feat, fc_w, fc_b, ln_g, ln_b, hfc, hfcpack);
    pack_whh_kernel<<<(64 * 8 * 32 * 16 + 255) / 256, 256, 0, stream>>>(w_hh, b_ih, b_hh, whhpack, bias);
    pack_wih_kernel<<<(Ln * 64 * 8 * 32 * 16 + 255) / 256, 256, 0, stream>>>(W, w_ih, wihpack);
    build_idx_kernel<<<(Bn * Ln * Nn + 255) / 256, 256, 0, stream>>>(adj, idxb);
    pmat_kernel<<<dim3(64, 4), 256, 0, stream>>>(hfcpack, wihpack, P);
    lstm_seq_kernel<<<SEQn / MT, 256, 0, stream>>>(whhpack, bias, P, idxb, hstate);
    final_kernel<<<(ROWS * Hn + 255) / 256, 256, 0, stream>>>(hfc, hstate, (float*)d_out);
}
